// EdgeRegressionNetWithMLP_71768903516435
// MI455X (gfx1250) — compile-verified, matrix operand order checked by reading, not run on hardware
//
#include <hip/hip_runtime.h>
#include <hip/hip_bf16.h>

// ---------------------------------------------------------------------------
// EdgeRegressionNetWithMLP fused CDNA5 implementation (gfx1250, wave32 WMMA)
//
// Precision: f16 WMMA (V_WMMA_F32_16X16X32_F16) with f32 accumulation.
// Pipeline per 16-edge tile (one wave owns a tile):
//   gather node_f16[src]/[dst] -> LDS combined[16x384]
//   edge_f = relu(edge_attr @ We + be)  (WMMA, into combined cols 256..383)
//   h1 = relu(combined @ W1 + b1)       (WMMA, 16x512 in LDS)
//   h2 = relu(h1 @ W2 + b2)             (WMMA, overlaid on combined)
//   out = h2 @ W3 + b3                  (VALU dot, W3 broadcast via LDS)
//
// B-reuse: 8 waves per workgroup walk the identical packed-weight stream and
// are phase-locked with __syncthreads() at each n-chunk boundary, so the
// 48KB-per-chunk B window is served out of WGP$ for 7 of 8 waves. This cuts
// the dominant L2 weight traffic from ~20.5 GB to ~2.6 GB per call, bringing
// it in balance with the ~338 GFLOP of f16 WMMA work.
// ---------------------------------------------------------------------------

typedef _Float16 f16;
typedef __attribute__((ext_vector_type(16))) _Float16 v16h;
typedef __attribute__((ext_vector_type(8)))  _Float16 v8h;
typedef __attribute__((ext_vector_type(4)))  _Float16 v4h;
typedef __attribute__((ext_vector_type(8)))  float    v8f;

#define N_NODES 50000
#define N_EDGES 500000
#define ETILES  (N_EDGES / 16)   // 31250
#define EWAVES  8                // waves (tiles) per workgroup

__device__ __forceinline__ v16h join16(v8h lo, v8h hi) {
    union { v16h v; v8h h[2]; } u;
    u.h[0] = lo; u.h[1] = hi;
    return u.v;
}

__device__ __forceinline__ v8f splat8(float x) {
    v8f a;
#pragma unroll
    for (int r = 0; r < 8; ++r) a[r] = x;
    return a;
}

__device__ __forceinline__ v8f wmma_f16(v16h a, v16h b, v8f c) {
    // (neg_a, A, neg_b, B, c_mod, C, reuse_a, reuse_b)
    return __builtin_amdgcn_wmma_f32_16x16x32_f16(false, a, false, b,
                                                  (short)0, c, false, false);
}

// ---------------------------------------------------------------------------
// Pack f32 weight matrix W[Ktot x Ntot] (row-major, fan_in x fan_out) into f16
// WMMA B-fragments. Fragment f = n_tile*kTiles + k_tile holds 32 lanes x 16
// halves. Element (l, i) of a fragment is W[K][N] with
//   K = kt*32 + (i>>3)*16 + (l>>4)*8 + (i&7),   N = nt*16 + (l&15)
// (ISA 7.12.2 16-bit B-matrix layout, wave32).
// ---------------------------------------------------------------------------
__global__ void pack_w_kernel(const float* __restrict__ W, f16* __restrict__ out,
                              int Ntot, int kTiles, int total) {
    for (int idx = blockIdx.x * blockDim.x + threadIdx.x; idx < total;
         idx += gridDim.x * blockDim.x) {
        int f  = idx >> 9;
        int r  = idx & 511;
        int l  = r >> 4;
        int i  = r & 15;
        int nt = f / kTiles;
        int kt = f - nt * kTiles;
        int K  = kt * 32 + ((i >> 3) << 4) + ((l >> 4) << 3) + (i & 7);
        int N  = nt * 16 + (l & 15);
        out[idx] = (f16)W[(size_t)K * Ntot + N];
    }
}

// ---------------------------------------------------------------------------
// node_f16 = (f16) relu(x @ Wn + bn).  One wave per 16-node tile.
// ---------------------------------------------------------------------------
__global__ void node_feat_kernel(const float* __restrict__ x,
                                 const float* __restrict__ bn,
                                 const f16*  __restrict__ Wnp,
                                 f16* __restrict__ nf) {
    __shared__ __align__(16) f16 xs_all[4][16 * 128];
    const int l  = threadIdx.x & 31;
    const int wv = threadIdx.x >> 5;
    const int tile = blockIdx.x * 4 + wv;
    if (tile >= N_NODES / 16) return;                 // wave-uniform
    f16* xs = xs_all[wv];
    const int n0 = tile * 16;
    const int m  = l & 15;
    const int h  = l >> 4;

    // Stage 16x128 f32 tile -> LDS f16 (64 elements per lane).
#pragma unroll
    for (int t = 0; t < 16; ++t) {
        int off = l * 64 + t * 4;
        int row = off >> 7, col = off & 127;
        const float4 v = *(const float4*)(x + (size_t)(n0 + row) * 128 + col);
        v4h p; p[0] = (f16)v.x; p[1] = (f16)v.y; p[2] = (f16)v.z; p[3] = (f16)v.w;
        *(v4h*)(xs + row * 128 + col) = p;
    }

    for (int nc = 0; nc < 2; ++nc) {                  // 2 chunks of 4 n-tiles
        v8f acc[4];
#pragma unroll
        for (int j = 0; j < 4; ++j) acc[j] = splat8(bn[(nc * 4 + j) * 16 + m]);
        for (int kt = 0; kt < 4; ++kt) {              // K = 128 -> 4 k-tiles
            int k0  = kt * 32 + h * 8;
            v16h a = join16(*(const v8h*)(xs + m * 128 + k0),
                            *(const v8h*)(xs + m * 128 + k0 + 16));
            v16h b[4];
#pragma unroll
            for (int j = 0; j < 4; ++j)
                b[j] = *(const v16h*)(Wnp + (size_t)(((nc * 4 + j) * 4 + kt) * 512) + l * 16);
#pragma unroll
            for (int j = 0; j < 4; ++j) acc[j] = wmma_f16(a, b[j], acc[j]);
        }
#pragma unroll
        for (int j = 0; j < 4; ++j)
#pragma unroll
            for (int r = 0; r < 8; ++r)
                nf[(size_t)(n0 + r + 8 * h) * 128 + (nc * 4 + j) * 16 + m] =
                    (f16)fmaxf(acc[j][r], 0.0f);
    }
}

// ---------------------------------------------------------------------------
// Fused edge pipeline: one wave per 16-edge tile, EWAVES waves per block,
// phase-locked so the shared packed-weight stream hits WGP$.
// Per-wave LDS: combined 16x392 f16 (12544B) + h1 16x520 f16 (16640B)
//               + ea 16x64 f16 (2048B, later overlaid by W3 f32) = 31232B.
// ---------------------------------------------------------------------------
#define CMB_STRIDE 392
#define H1_STRIDE  520
#define WAVE_LDS   31232

__global__ void edge_fused_kernel(const int*   __restrict__ eidx,
                                  const float* __restrict__ eattr,
                                  const float* __restrict__ be,
                                  const float* __restrict__ b1,
                                  const float* __restrict__ b2,
                                  const float* __restrict__ W3,
                                  const float* __restrict__ b3,
                                  const f16*   __restrict__ Wep,
                                  const f16*   __restrict__ W1p,
                                  const f16*   __restrict__ W2p,
                                  const f16*   __restrict__ nf,
                                  float* __restrict__ out) {
    extern __shared__ __align__(16) char smem[];
    const int l  = threadIdx.x & 31;
    const int wv = threadIdx.x >> 5;
    const int m  = l & 15;
    const int h  = l >> 4;
    f16*   cmb = (f16*)(smem + wv * WAVE_LDS);
    f16*   h1  = cmb + 16 * CMB_STRIDE;               // +6272 halves
    f16*   ea  = h1 + 16 * H1_STRIDE;                 // +8320 halves
    float* w3l = (float*)ea;                          // overlay, used after ea

    const int  tile  = blockIdx.x * EWAVES + wv;
    const bool valid = (tile < ETILES);
    const int  e0    = valid ? tile * 16 : 0;  // invalid waves shadow tile 0,
                                               // keep uniform barrier flow,
                                               // suppress the final store.

    // ---- Gather: lane l copies node row for edge m, endpoint h (src/dst).
    const int node = eidx[h * N_EDGES + e0 + m];
    const f16* nrow = nf + (size_t)node * 128;
#pragma unroll
    for (int t = 0; t < 16; ++t)
        *(v8h*)(cmb + m * CMB_STRIDE + h * 128 + t * 8) = *(const v8h*)(nrow + t * 8);

    // ---- Stage edge_attr 16x64 f32 -> LDS f16.
#pragma unroll
    for (int t = 0; t < 8; ++t) {
        int off = l * 32 + t * 4;
        int row = off >> 6, col = off & 63;
        const float4 v = *(const float4*)(eattr + (size_t)(e0 + row) * 64 + col);
        v4h p; p[0] = (f16)v.x; p[1] = (f16)v.y; p[2] = (f16)v.z; p[3] = (f16)v.w;
        *(v4h*)(ea + row * 64 + col) = p;
    }

    // ---- edge_f = relu(ea @ We + be) -> combined cols 256..383.
    for (int nt = 0; nt < 8; ++nt) {
        v8f acc = splat8(be[nt * 16 + m]);
#pragma unroll
        for (int kt = 0; kt < 2; ++kt) {
            int k0 = kt * 32 + h * 8;
            v16h a = join16(*(const v8h*)(ea + m * 64 + k0),
                            *(const v8h*)(ea + m * 64 + k0 + 16));
            v16h b = *(const v16h*)(Wep + (size_t)((nt * 2 + kt) * 512) + l * 16);
            acc = wmma_f16(a, b, acc);
        }
#pragma unroll
        for (int r = 0; r < 8; ++r)
            cmb[(r + 8 * h) * CMB_STRIDE + 256 + nt * 16 + m] = (f16)fmaxf(acc[r], 0.0f);
    }

    // ---- Broadcast W3 into LDS (ea region is now free; per-wave LDS in-order).
#pragma unroll
    for (int t = 0; t < 8; ++t) w3l[t * 32 + l] = W3[t * 32 + l];

    __syncthreads();   // phase-lock all waves before the shared weight stream

    // ---- Layer 1: h1 = relu(combined[16x384] @ W1 + b1), N = 512.
    for (int nc = 0; nc < 8; ++nc) {                  // 8 chunks of 4 n-tiles
        if (nc < 7)    // warm WGP$/L2 for the next 48KB chunk (global_prefetch)
            __builtin_prefetch(W1p + (size_t)(((nc + 1) * 4) * 12) * 512 + l * 16, 0, 1);
        v8f acc[4];
#pragma unroll
        for (int j = 0; j < 4; ++j) acc[j] = splat8(b1[(nc * 4 + j) * 16 + m]);
        for (int kt = 0; kt < 12; ++kt) {             // K = 384
            int k0 = kt * 32 + h * 8;
            v16h a = join16(*(const v8h*)(cmb + m * CMB_STRIDE + k0),
                            *(const v8h*)(cmb + m * CMB_STRIDE + k0 + 16));
            v16h b[4];
#pragma unroll
            for (int j = 0; j < 4; ++j)
                b[j] = *(const v16h*)(W1p + (size_t)(((nc * 4 + j) * 12 + kt) * 512) + l * 16);
#pragma unroll
            for (int j = 0; j < 4; ++j) acc[j] = wmma_f16(a, b[j], acc[j]);
        }
#pragma unroll
        for (int j = 0; j < 4; ++j)
#pragma unroll
            for (int r = 0; r < 8; ++r)
                h1[(r + 8 * h) * H1_STRIDE + (nc * 4 + j) * 16 + m] =
                    (f16)fmaxf(acc[j][r], 0.0f);
        __syncthreads();  // keep all waves inside the same WGP$ weight window
    }

    // ---- Layer 2: h2 = relu(h1[16x512] @ W2 + b2), N = 256; h2 -> cmb cols 0..255.
    for (int nc = 0; nc < 4; ++nc) {
        if (nc < 3)
            __builtin_prefetch(W2p + (size_t)(((nc + 1) * 4) * 16) * 512 + l * 16, 0, 1);
        v8f acc[4];
#pragma unroll
        for (int j = 0; j < 4; ++j) acc[j] = splat8(b2[(nc * 4 + j) * 16 + m]);
        for (int kt = 0; kt < 16; ++kt) {             // K = 512
            int k0 = kt * 32 + h * 8;
            v16h a = join16(*(const v8h*)(h1 + m * H1_STRIDE + k0),
                            *(const v8h*)(h1 + m * H1_STRIDE + k0 + 16));
            v16h b[4];
#pragma unroll
            for (int j = 0; j < 4; ++j)
                b[j] = *(const v16h*)(W2p + (size_t)(((nc * 4 + j) * 16 + kt) * 512) + l * 16);
#pragma unroll
            for (int j = 0; j < 4; ++j) acc[j] = wmma_f16(a, b[j], acc[j]);
        }
#pragma unroll
        for (int j = 0; j < 4; ++j)
#pragma unroll
            for (int r = 0; r < 8; ++r)
                cmb[(r + 8 * h) * CMB_STRIDE + (nc * 4 + j) * 16 + m] =
                    (f16)fmaxf(acc[j][r], 0.0f);
        __syncthreads();
    }

    // ---- Head: out[e] = h2[e] . W3 + b3  (lanes 0..15, one edge each).
    if (valid && h == 0) {
        float s = b3[0];
        for (int t = 0; t < 32; ++t) {
            v8h c = *(const v8h*)(cmb + m * CMB_STRIDE + t * 8);
#pragma unroll
            for (int u = 0; u < 8; ++u) s += (float)c[u] * w3l[t * 8 + u];
        }
        out[e0 + m] = s;
    }
}

// ---------------------------------------------------------------------------
// Workspace layout (bytes): packed f16 weights then node_f16 table (~13.5 MB).
// ---------------------------------------------------------------------------
#define WS_WN 0
#define WS_WE (WS_WN + 16384 * 2)     // 32768
#define WS_W1 (WS_WE + 8192 * 2)      // 49152
#define WS_W2 (WS_W1 + 196608 * 2)    // 442368
#define WS_NF (WS_W2 + 131072 * 2)    // 704512

extern "C" void kernel_launch(void* const* d_in, const int* in_sizes, int n_in,
                              void* d_out, int out_size, void* d_ws, size_t ws_size,
                              hipStream_t stream) {
    (void)in_sizes; (void)n_in; (void)out_size; (void)ws_size;
    const float* x     = (const float*)d_in[0];
    const int*   eidx  = (const int*)  d_in[1];
    const float* eattr = (const float*)d_in[2];
    const float* Wn    = (const float*)d_in[3];
    const float* bn    = (const float*)d_in[4];
    const float* We    = (const float*)d_in[5];
    const float* be    = (const float*)d_in[6];
    const float* W1    = (const float*)d_in[7];
    const float* b1    = (const float*)d_in[8];
    const float* W2    = (const float*)d_in[9];
    const float* b2    = (const float*)d_in[10];
    const float* W3    = (const float*)d_in[11];
    const float* b3    = (const float*)d_in[12];

    char* ws  = (char*)d_ws;
    f16* Wnp = (f16*)(ws + WS_WN);
    f16* Wep = (f16*)(ws + WS_WE);
    f16* W1p = (f16*)(ws + WS_W1);
    f16* W2p = (f16*)(ws + WS_W2);
    f16* nf  = (f16*)(ws + WS_NF);

    // Allow >64KB dynamic LDS for the edge kernel (320KB/WGP on gfx1250).
    (void)hipFuncSetAttribute((const void*)edge_fused_kernel,
                              hipFuncAttributeMaxDynamicSharedMemorySize,
                              EWAVES * WAVE_LDS);

    // Pre-swizzle weights into WMMA B-fragment order (f16).
    pack_w_kernel<<<64,  256, 0, stream>>>(Wn, Wnp, 128, 4,  16384);
    pack_w_kernel<<<32,  256, 0, stream>>>(We, Wep, 128, 2,  8192);
    pack_w_kernel<<<768, 256, 0, stream>>>(W1, W1p, 512, 12, 196608);
    pack_w_kernel<<<512, 256, 0, stream>>>(W2, W2p, 256, 16, 131072);

    // Node features (3125 tiles, 4 waves per 128-thread block).
    node_feat_kernel<<<(3125 + 3) / 4, 128, 0, stream>>>(x, bn, Wnp, nf);

    // Fused edge pipeline (31250 tiles, 8 phase-locked waves per block).
    edge_fused_kernel<<<(ETILES + EWAVES - 1) / EWAVES, 32 * EWAVES,
                        EWAVES * WAVE_LDS, stream>>>(
        eidx, eattr, be, b1, b2, W3, b3, Wep, W1p, W2p, nf, (float*)d_out);
}
